// GraphHead_64003602645421
// MI455X (gfx1250) — compile-verified, matrix operand order checked by reading, not run on hardware
//
#include <hip/hip_runtime.h>
#include <hip/hip_bf16.h>
#include <stdint.h>

#define BS 64
#define SEQ 576
#define FT_IN 768
#define HID 512
#define NNODE 577
#define NUM_NODES (BS*NNODE)   // 36928
#define LEAF_TOTAL (BS*SEQ)    // 36864

typedef __attribute__((ext_vector_type(16))) __bf16 v16bf;
typedef __attribute__((ext_vector_type(8)))  float  v8f;

__device__ __forceinline__ unsigned f2bf1(float a){
  unsigned u = __float_as_uint(a);
  return (u + 0x7FFFu + ((u >> 16) & 1u)) >> 16;
}
// RNE round both floats to bf16, pack into one dword using 32-bit ops only
__device__ __forceinline__ unsigned bf16pk(float a, float b){
  unsigned ua = __float_as_uint(a), ub = __float_as_uint(b);
  ua = (ua + 0x7FFFu + ((ua >> 16) & 1u)) >> 16;
  ub =  ub + 0x7FFFu + ((ub >> 16) & 1u);
  return ua | (ub & 0xFFFF0000u);
}
__device__ __forceinline__ float geluf(float x){
  float x3 = x*x*x;
  return 0.5f*x*(1.0f + tanhf(0.7978845608028654f*(x + 0.044715f*x3)));
}
__device__ __forceinline__ float lrelu(float x){
  return fmaxf(x, 0.0f) + 0.2f*fminf(x, 0.0f);
}

// ---------------------------------------------------------------------------
// Weight fp32 [K x N] row-major -> bf16 [N x K] (transposed) for B fragments
// ---------------------------------------------------------------------------
__global__ void k_wt_bf16(const float* __restrict__ in, unsigned short* __restrict__ out,
                          int K, int N){
  int i = blockIdx.x*256 + threadIdx.x;
  if (i >= K*N) return;
  int n = i / K, k = i - n*K;
  out[i] = (unsigned short)f2bf1(in[(size_t)k*N + n]);
}

__global__ void k_cat2(const float* __restrict__ a, const float* __restrict__ b,
                       float* __restrict__ o){
  int i = blockIdx.x*256 + threadIdx.x;   // 1024
  o[i] = (i < HID) ? a[i] : b[i - HID];
}

__global__ void k_fill_cls(const float* __restrict__ pooled, float* __restrict__ x){
  int i = blockIdx.x*256 + threadIdx.x;   // BS*HID
  int b = i >> 9, c = i & (HID-1);
  x[(size_t)b*NNODE*HID + c] = pooled[i];
}

// ---------------------------------------------------------------------------
// WMMA bf16 GEMM: C[M,N] = act(A[M,K] @ B[K,N] + bias) (+res) ; B as [N,K] bf16
// 128x128x32 block tile, 512 threads = 16 waves in a 4x4 grid, each wave owns a
// 32x32 sub-tile (2x2 WMMA tiles). Low per-wave register pressure (~100 VGPRs:
// 32 acc + 32 frag + 12 staged) -> no scratch spills, 4 waves/SIMD latency
// hiding. LDS double-buffered; next k-tile's global loads are in flight across
// the WMMA section. OOB rows handled by address clamping (garbage rows map
// only to masked-out C rows).
// ---------------------------------------------------------------------------
#define BM 128
#define BN 128
#define BK 32
#define LDSP 40   // padded LDS row pitch (bf16 elems): 80B rows, 16B aligned

union Frag { v16bf v; uint4 q[2]; };

__global__ __launch_bounds__(512, 1)
void k_gemm_bf16(const float* __restrict__ A, size_t lda,
                 const unsigned short* __restrict__ Bt,   // [N x K] bf16
                 const float* __restrict__ bias,
                 float* __restrict__ C, size_t ldc,
                 int M, int N, int K,
                 int act, int remap576, const float* __restrict__ res)
{
  __shared__ __align__(16) unsigned short lA[2][BM*LDSP];
  __shared__ __align__(16) unsigned short lB[2][BN*LDSP];

  const int t    = threadIdx.x;                   // 0..511
  const int wave = t >> 5, lane = t & 31;
  const int wm   = wave & 3, wn = wave >> 2;      // 4x4 wave grid
  const int idx  = lane & 15, half = lane >> 4;   // fragment lane coords
  const int bm0  = blockIdx.y * BM;
  const int bn0  = blockIdx.x * BN;

  // staging slot coordinates (fixed per thread)
  int arow[2], acol[2];
#pragma unroll
  for (int i = 0; i < 2; i++){ int id = t + 512*i; arow[i] = id >> 3; acol[i] = id & 7; }
  const int brow = t >> 2, bcol = t & 3;

  // clamped global offsets (branchless OOB handling)
  size_t aoff[2], boff;
#pragma unroll
  for (int i = 0; i < 2; i++){
    int gr = bm0 + arow[i]; if (gr > M-1) gr = M-1;
    aoff[i] = (size_t)gr*lda + acol[i]*4;
  }
  {
    int gn = bn0 + brow; if (gn > N-1) gn = N-1;
    boff = (size_t)gn*K + bcol*8;
  }

  v8f acc[2][2];
  v8f zero = {0.f,0.f,0.f,0.f,0.f,0.f,0.f,0.f};
#pragma unroll
  for (int i = 0; i < 2; i++)
#pragma unroll
    for (int j = 0; j < 2; j++) acc[i][j] = zero;

  float4 av[2]; uint4 bv;

  auto load_tiles = [&](int k0){
#pragma unroll
    for (int i = 0; i < 2; i++) av[i] = *(const float4*)(A + aoff[i] + k0);
    bv = *(const uint4*)(Bt + boff + k0);
  };
  auto store_tiles = [&](int buf){
#pragma unroll
    for (int i = 0; i < 2; i++){
      uint2 p;
      p.x = bf16pk(av[i].x, av[i].y);
      p.y = bf16pk(av[i].z, av[i].w);
      *(uint2*)&lA[buf][arow[i]*LDSP + acol[i]*4] = p;
    }
    *(uint4*)&lB[buf][brow*LDSP + bcol*8] = bv;
  };

  // prologue: fill buffer 0
  load_tiles(0);
  store_tiles(0);

  int buf = 0;
  for (int k0 = 0; k0 < K; k0 += BK){
    __syncthreads();                       // lA/lB[buf] visible to all waves
    const bool more = (k0 + BK) < K;
    if (more){
      load_tiles(k0 + BK);                 // in flight during WMMAs below
      if (k0 + 2*BK < K)
        __builtin_prefetch(A + aoff[0] + k0 + 2*BK, 0, 1);  // global_prefetch_b8
    }

    // ---- fragments (2x ds_load_b128 each, matches 16-bit A/B VGPR layout) --
    Frag af[2], bf[2];
#pragma unroll
    for (int mi = 0; mi < 2; mi++){
      const unsigned short* p = &lA[buf][(wm*32 + mi*16 + idx)*LDSP + half*8];
      af[mi].q[0] = *(const uint4*)p;        // K 0..7   of lane-half
      af[mi].q[1] = *(const uint4*)(p + 16); // K 16..23 of lane-half
    }
#pragma unroll
    for (int ni = 0; ni < 2; ni++){
      const unsigned short* p = &lB[buf][(wn*32 + ni*16 + idx)*LDSP + half*8];
      bf[ni].q[0] = *(const uint4*)p;
      bf[ni].q[1] = *(const uint4*)(p + 16);
    }
#pragma unroll
    for (int mi = 0; mi < 2; mi++)
#pragma unroll
      for (int ni = 0; ni < 2; ni++)
        acc[mi][ni] = __builtin_amdgcn_wmma_f32_16x16x32_bf16(
            false, af[mi].v, false, bf[ni].v, (short)0, acc[mi][ni], false, false);

    if (more) store_tiles(buf ^ 1);        // write other buffer; no conflict
    buf ^= 1;
  }

  // ---- epilogue: C layout — VGPR v, lanes 0-15: M=v, lanes 16-31: M=v+8 ----
#pragma unroll
  for (int mi = 0; mi < 2; mi++){
#pragma unroll
    for (int ni = 0; ni < 2; ni++){
      int col = bn0 + wn*32 + ni*16 + idx;
      if (col >= N) continue;
      float bz = bias ? bias[col] : 0.0f;
#pragma unroll
      for (int v = 0; v < 8; v++){
        int row = bm0 + wm*32 + mi*16 + half*8 + v;
        if (row >= M) continue;
        float val = acc[mi][ni][v] + bz;
        if (act) val = geluf(val);
        if (res) val += res[(size_t)row*N + col];
        size_t orow = (size_t)row;
        if (remap576) orow = (size_t)row + (size_t)(row/576) + 1; // insert CLS rows
        C[orow*ldc + col] = val;
      }
    }
  }
}

// ---------------------------------------------------------------------------
// GATv2 leaf nodes: each leaf has exactly 2 in-edges (hub->leaf, self-loop)
// xlr: [NUM_NODES x ld], xl at col 0, xr at col 512
// ---------------------------------------------------------------------------
__global__ __launch_bounds__(256)
void k_gat_leaf(const float* __restrict__ xlr, int ld,
                const float* __restrict__ att, const float* __restrict__ bias,
                float* __restrict__ out, int heads, int act)
{
  __shared__ float s0[HID], s1[HID];
  __shared__ float a0s[4], a1s[4];
  const int t = threadIdx.x;
  const int blk = blockIdx.x;
  const int b = blk / SEQ, j = blk - b*SEQ;
  const size_t g0 = (size_t)b*NNODE;
  const size_t g  = g0 + 1 + j;
  const int chead = HID / heads;

  float xl0[2], xlg[2];
#pragma unroll
  for (int i = 0; i < 2; i++){
    int c = t + 256*i;
    float l0 = xlr[g0*ld + c];
    float lg = xlr[g*ld + c];
    float rg = xlr[g*ld + HID + c];
    float ac = att[c];
    xl0[i] = l0; xlg[i] = lg;
    s0[c] = ac * lrelu(l0 + rg);   // edge hub->leaf
    s1[c] = ac * lrelu(lg + rg);   // self loop
  }
  __syncthreads();
  for (int off = chead >> 1; off > 0; off >>= 1){
#pragma unroll
    for (int i = 0; i < 2; i++){
      int c = t + 256*i;
      if ((c & (chead-1)) < off){ s0[c] += s0[c+off]; s1[c] += s1[c+off]; }
    }
    __syncthreads();
  }
  if (t < heads){
    float e0 = s0[t*chead], e1 = s1[t*chead];
    float m  = fmaxf(e0, e1);
    float p0 = expf(e0 - m), p1 = expf(e1 - m);
    float inv = 1.0f / (p0 + p1 + 1e-16f);
    a0s[t] = p0*inv; a1s[t] = p1*inv;
  }
  __syncthreads();
#pragma unroll
  for (int i = 0; i < 2; i++){
    int c = t + 256*i;
    int h = c / chead;
    float v = a0s[h]*xl0[i] + a1s[h]*xlg[i] + bias[c];
    if (act) v = geluf(v);
    out[g*HID + c] = v;
  }
}

// ---------------------------------------------------------------------------
// GATv2 hub node: 577 in-edges (all leaves + self loop). One block per batch.
// ---------------------------------------------------------------------------
__global__ __launch_bounds__(256)
void k_gat_hub(const float* __restrict__ xlr, int ld,
               const float* __restrict__ att, const float* __restrict__ bias,
               float* __restrict__ out, int heads, int act)
{
  __shared__ float xr_s[HID];
  __shared__ float att_s[HID];
  __shared__ float xls[8*HID];
  __shared__ float e_s[NNODE*4 + 4];
  __shared__ float inv_s[4];

  const int t = threadIdx.x;
  const int b = blockIdx.x;
  const size_t g0 = (size_t)b*NNODE;
  const int chead = HID / heads;
  const int w = t >> 5, lane = t & 31;

#pragma unroll
  for (int i = 0; i < 2; i++){
    int c = t + 256*i;
    xr_s[c]  = xlr[g0*ld + HID + c];
    att_s[c] = att[c];
  }
  __syncthreads();

  // edge scores e_j,h for j=0..576 (wave per row, shfl reduce)
  for (int j0 = 0; j0 < NNODE; j0 += 8){
#pragma unroll
    for (int i = 0; i < 16; i++){
      int id = t + 256*i;
      int r = id >> 9, c = id & (HID-1);
      int j = j0 + r;
      xls[id] = (j < NNODE) ? xlr[(g0 + j)*ld + c] : 0.0f;
    }
    __syncthreads();
    int j = j0 + w;
    if (j < NNODE){
      float a0=0.f, a1=0.f, a2=0.f, a3=0.f;
      if (heads == 4){
#pragma unroll
        for (int i = 0; i < 16; i++){
          int c = lane + 32*i;
          float v = att_s[c]*lrelu(xls[w*HID + c] + xr_s[c]);
          if (i < 4) a0 += v; else if (i < 8) a1 += v;
          else if (i < 12) a2 += v; else a3 += v;
        }
      } else {
#pragma unroll
        for (int i = 0; i < 16; i++){
          int c = lane + 32*i;
          a0 += att_s[c]*lrelu(xls[w*HID + c] + xr_s[c]);
        }
      }
#pragma unroll
      for (int off = 16; off > 0; off >>= 1){
        a0 += __shfl_xor(a0, off, 32);
        a1 += __shfl_xor(a1, off, 32);
        a2 += __shfl_xor(a2, off, 32);
        a3 += __shfl_xor(a3, off, 32);
      }
      if (lane == 0){
        e_s[j*4+0] = a0; e_s[j*4+1] = a1; e_s[j*4+2] = a2; e_s[j*4+3] = a3;
      }
    }
    __syncthreads();
  }
  // softmax stats per head (serial, tiny)
  if (t < heads){
    float m = -1e30f;
    for (int j = 0; j < NNODE; j++) m = fmaxf(m, e_s[j*4+t]);
    float s = 0.f;
    for (int j = 0; j < NNODE; j++){ float p = expf(e_s[j*4+t] - m); e_s[j*4+t] = p; s += p; }
    inv_s[t] = 1.0f / (s + 1e-16f);
  }
  __syncthreads();
  // weighted aggregation of xl rows (coalesced across threads)
#pragma unroll
  for (int i = 0; i < 2; i++){
    int c = t + 256*i;
    int h = c / chead;
    float acc = 0.f;
    for (int j = 0; j < NNODE; j++)
      acc += e_s[j*4+h] * xlr[(g0 + j)*ld + c];
    float v = acc*inv_s[h] + bias[c];
    if (act) v = geluf(v);
    out[g0*HID + c] = v;
  }
}

__global__ __launch_bounds__(256)
void k_mean(const float* __restrict__ h2, float* __restrict__ o){
  int t = threadIdx.x, b = blockIdx.x;
#pragma unroll
  for (int i = 0; i < 2; i++){
    int c = t + 256*i;
    float s = 0.f;
    for (int j = 0; j < NNODE; j++) s += h2[((size_t)b*NNODE + j)*HID + c];
    o[(size_t)b*HID + c] = s * (1.0f/(float)NNODE);
  }
}

// ---------------------------------------------------------------------------
extern "C" void kernel_launch(void* const* d_in, const int* in_sizes, int n_in,
                              void* d_out, int out_size, void* d_ws, size_t ws_size,
                              hipStream_t stream)
{
  (void)in_sizes; (void)n_in; (void)out_size; (void)ws_size;
  const float* hs    = (const float*)d_in[0];
  const float* pooled= (const float*)d_in[1];
  const float* pw1   = (const float*)d_in[2];
  const float* pb1   = (const float*)d_in[3];
  const float* pw2   = (const float*)d_in[4];
  const float* pb2   = (const float*)d_in[5];
  const float* wl1   = (const float*)d_in[6];
  const float* bl1   = (const float*)d_in[7];
  const float* wr1   = (const float*)d_in[8];
  const float* br1   = (const float*)d_in[9];
  const float* att1  = (const float*)d_in[10];
  const float* bias1 = (const float*)d_in[11];
  const float* wl2   = (const float*)d_in[12];
  const float* bl2   = (const float*)d_in[13];
  const float* wr2   = (const float*)d_in[14];
  const float* br2   = (const float*)d_in[15];
  const float* att2  = (const float*)d_in[16];
  const float* bias2 = (const float*)d_in[17];
  const float* linw  = (const float*)d_in[18];
  const float* linb  = (const float*)d_in[19];
  float* out = (float*)d_out;

  char* ws = (char*)d_ws;
  size_t off = 0;
  auto alloc = [&](size_t bytes)->char* {
    char* p = ws + off; off += (bytes + 255) & ~(size_t)255; return p;
  };
  float* buf_x  = (float*)alloc((size_t)NUM_NODES*HID*4);     // x / h2
  float* buf_lr = (float*)alloc((size_t)NUM_NODES*1024*4);    // h_mid / xl|xr
  float* buf_h  = (float*)alloc((size_t)NUM_NODES*HID*4);     // h1
  unsigned short* wt_pw1 = (unsigned short*)alloc((size_t)HID*FT_IN*2);
  unsigned short* wt_pw2 = (unsigned short*)alloc((size_t)HID*HID*2);
  unsigned short* wt_lr1 = (unsigned short*)alloc((size_t)1024*HID*2);
  unsigned short* wt_lr2 = (unsigned short*)alloc((size_t)1024*HID*2);
  unsigned short* wt_lin = (unsigned short*)alloc((size_t)HID*HID*2);
  float* blr1 = (float*)alloc(1024*4);
  float* blr2 = (float*)alloc(1024*4);

  // ---- one-time weight prep (bf16, transposed) ----
  k_wt_bf16<<<(FT_IN*HID+255)/256,256,0,stream>>>(pw1, wt_pw1, FT_IN, HID);
  k_wt_bf16<<<(HID*HID+255)/256,256,0,stream>>>(pw2, wt_pw2, HID, HID);
  k_wt_bf16<<<(HID*HID+255)/256,256,0,stream>>>(wl1, wt_lr1,                  HID, HID);
  k_wt_bf16<<<(HID*HID+255)/256,256,0,stream>>>(wr1, wt_lr1 + (size_t)HID*HID, HID, HID);
  k_wt_bf16<<<(HID*HID+255)/256,256,0,stream>>>(wl2, wt_lr2,                  HID, HID);
  k_wt_bf16<<<(HID*HID+255)/256,256,0,stream>>>(wr2, wt_lr2 + (size_t)HID*HID, HID, HID);
  k_wt_bf16<<<(HID*HID+255)/256,256,0,stream>>>(linw, wt_lin, HID, HID);
  k_cat2<<<4,256,0,stream>>>(bl1, br1, blr1);
  k_cat2<<<4,256,0,stream>>>(bl2, br2, blr2);
  k_fill_cls<<<(BS*HID)/256,256,0,stream>>>(pooled, buf_x);

  // ---- MLP stage: h = gelu(hs@pw1+pb1) ; x_leaf = h@pw2+pb2 (remapped) ----
  k_gemm_bf16<<<dim3(HID/128, (LEAF_TOTAL+127)/128), 512, 0, stream>>>(
      hs, (size_t)FT_IN, wt_pw1, pb1, buf_lr, (size_t)HID,
      LEAF_TOTAL, HID, FT_IN, /*gelu*/1, 0, nullptr);
  k_gemm_bf16<<<dim3(HID/128, (LEAF_TOTAL+127)/128), 512, 0, stream>>>(
      buf_lr, (size_t)HID, wt_pw2, pb2, buf_x, (size_t)HID,
      LEAF_TOTAL, HID, HID, 0, /*remap576*/1, nullptr);

  // ---- GAT layer 1 ----
  k_gemm_bf16<<<dim3(1024/128, (NUM_NODES+127)/128), 512, 0, stream>>>(
      buf_x, (size_t)HID, wt_lr1, blr1, buf_lr, (size_t)1024,
      NUM_NODES, 1024, HID, 0, 0, nullptr);
  k_gat_leaf<<<LEAF_TOTAL, 256, 0, stream>>>(buf_lr, 1024, att1, bias1, buf_h, 4, 1);
  k_gat_hub<<<BS, 256, 0, stream>>>(buf_lr, 1024, att1, bias1, buf_h, 4, 1);

  // ---- GAT layer 2 ----
  k_gemm_bf16<<<dim3(1024/128, (NUM_NODES+127)/128), 512, 0, stream>>>(
      buf_h, (size_t)HID, wt_lr2, blr2, buf_lr, (size_t)1024,
      NUM_NODES, 1024, HID, 0, 0, nullptr);
  k_gat_leaf<<<LEAF_TOTAL, 256, 0, stream>>>(buf_lr, 1024, att2, bias2, buf_x, 1, 0);
  k_gat_hub<<<BS, 256, 0, stream>>>(buf_lr, 1024, att2, bias2, buf_x, 1, 0);

  // ---- outputs: graph_mean and cls @ linw + linb + pooled ----
  k_mean<<<BS, 256, 0, stream>>>(buf_x, out + (size_t)BS*HID);
  k_gemm_bf16<<<dim3(HID/128, 1), 512, 0, stream>>>(
      buf_x, (size_t)NNODE*HID, wt_lin, linb, out, (size_t)HID,
      BS, HID, HID, 0, 0, pooled);
}
